// GridSelfAttention_32976758899127
// MI455X (gfx1250) — matrix operand order repaired and checked by reading, not compiled, not run on hardware
//
#include <hip/hip_runtime.h>

#define C_DIM 256
#define N_TOK 4096
#define B_SZ  4
#define TILE  64
#define NT    (N_TOK / TILE)
#define LDQ   272   // C + 16 bf16 elems -> 544B row stride (16B aligned)
#define LDVT  80    // 64 + 16 bf16 elems -> 160B
#define LD_S  68    // f32 S tile row stride (272B, 16B aligned)
#define LDP   80    // bf16 P tile row stride

typedef __bf16 bf16_t;
typedef __attribute__((ext_vector_type(16))) __bf16 bf16x16;
typedef __attribute__((ext_vector_type(8)))  __bf16 bf16x8;
typedef __attribute__((ext_vector_type(8)))  float  f32x8;
typedef __attribute__((ext_vector_type(4)))  int    i32x4;

typedef __attribute__((address_space(1))) i32x4 gas_i32x4;
typedef __attribute__((address_space(3))) i32x4 las_i32x4;

#if defined(__gfx1250__) && __has_builtin(__builtin_amdgcn_global_load_async_to_lds_b128)
#define HAVE_ASYNC_LDS 1
#else
#define HAVE_ASYNC_LDS 0
#endif

// 16-byte global -> LDS copy; async (ASYNCcnt) when available, sync fallback.
__device__ __forceinline__ void copy16_g2l(const bf16_t* gptr, bf16_t* lptr) {
#if HAVE_ASYNC_LDS
  __builtin_amdgcn_global_load_async_to_lds_b128(
      (gas_i32x4*)gptr, (las_i32x4*)lptr, 0, 0);
#else
  *(bf16x8*)lptr = *(const bf16x8*)gptr;
#endif
}

// Wait until at most 16 (just-issued batch) / 0 async copies remain outstanding.
__device__ __forceinline__ void wait_async_le16(bool keep16) {
#if HAVE_ASYNC_LDS
#if __has_builtin(__builtin_amdgcn_s_wait_asynccnt)
  if (keep16) __builtin_amdgcn_s_wait_asynccnt(16);
  else        __builtin_amdgcn_s_wait_asynccnt(0);
#else
  if (keep16) asm volatile("s_wait_asynccnt 16" ::: "memory");
  else        asm volatile("s_wait_asynccnt 0"  ::: "memory");
#endif
#endif
}

// A-matrix 16x32 bf16 fragment (ISA 7.12.2): lane = row (lane&15);
// lanes 0-15 hold K = 0..7 (v0..v3) and 16..23 (v4..v7); lanes 16-31 hold K+8.
__device__ __forceinline__ bf16x16 load_fragA(const bf16_t* base, int row, int ld,
                                              int k0, int lane) {
  const int half = (lane >> 4) & 1;
  const bf16_t* p = base + row * ld + k0 + half * 8;
  bf16x8 lo = *(const bf16x8*)(p);
  bf16x8 hi = *(const bf16x8*)(p + 16);
  bf16x16 f;
#pragma unroll
  for (int i = 0; i < 8; ++i) { f[i] = lo[i]; f[8 + i] = hi[i]; }
  return f;
}

// B-matrix 32x16 bf16 fragment: lane = column (lane&15); lanes 0-15 hold
// K = 0..7 and 16..23, lanes 16-31 hold K+8 (same order as fragA).
__device__ __forceinline__ bf16x16 load_fragB(const bf16_t* base, int col, int ld,
                                              int k0, int lane) {
  const int half = (lane >> 4) & 1;
  const bf16_t* p = base + col * ld + k0 + half * 8;
  bf16x8 lo = *(const bf16x8*)(p);
  bf16x8 hi = *(const bf16x8*)(p + 16);
  bf16x16 f;
#pragma unroll
  for (int i = 0; i < 8; ++i) { f[i] = lo[i]; f[8 + i] = hi[i]; }
  return f;
}

__device__ __forceinline__ f32x8 wmma_bf16(bf16x16 a, bf16x16 b, f32x8 c) {
  return __builtin_amdgcn_wmma_f32_16x16x32_bf16(false, a, false, b,
                                                 (short)0, c, false, false);
}

// ---------------------------------------------------------------- kernel 0
__global__ void __launch_bounds__(256)
prep_weights(const float* __restrict__ Wq, const float* __restrict__ Wk,
             const float* __restrict__ Wv, bf16_t* __restrict__ wAll) {
  int idx = blockIdx.x * blockDim.x + threadIdx.x;
  const int per = C_DIM * C_DIM;
  if (idx >= 3 * per) return;
  int mat = idx / per;
  int rem = idx - mat * per;
  const float* src = (mat == 0) ? Wq : (mat == 1) ? Wk : Wv;
  wAll[idx] = (bf16_t)src[rem];
}

// ---------------------------------------------------------------- kernel 1
// Q/K -> [B][N][C] bf16 ; V -> transposed [B][C][N] bf16
__global__ void __launch_bounds__(256)
qkv_kernel(const float* __restrict__ x, const bf16_t* __restrict__ wAll,
           const float* __restrict__ bq, const float* __restrict__ bk,
           const float* __restrict__ bv, bf16_t* __restrict__ qws,
           bf16_t* __restrict__ kws, bf16_t* __restrict__ vtws) {
  __shared__ bf16_t sX[TILE * LDQ];   // xf tile [token][c']

  const int bid = blockIdx.x;
  const int b   = bid >> 6;
  const int n0  = (bid & 63) * TILE;
  const int tid  = threadIdx.x;
  const int lane = tid & 31;
  const int wave = tid >> 5;
  const int half = lane >> 4;
  const int ln   = lane & 15;

  for (int idx = tid; idx < TILE * C_DIM; idx += 256) {
    int c = idx >> 6;
    int i = idx & 63;
    float v = x[(size_t)(b * C_DIM + c) * N_TOK + n0 + i];
    sX[i * LDQ + c] = (bf16_t)v;
  }
  __syncthreads();

  for (int tt = wave; tt < 192; tt += 8) {
    int mat = tt >> 6;
    int r   = (tt >> 4) & 3;
    int ct  = tt & 15;
    const bf16_t* wmat = wAll + (size_t)mat * C_DIM * C_DIM;
    int c = ct * 16 + ln;

    f32x8 acc = {};
#pragma unroll
    for (int kk = 0; kk < 8; ++kk) {
      bf16x16 a  = load_fragA(sX, r * 16 + ln, LDQ, kk * 32, lane);
      bf16x16 bf = load_fragB(wmat, c, C_DIM, kk * 32, lane);
      acc = wmma_bf16(a, bf, acc);
    }
    const float* bias = (mat == 0) ? bq : (mat == 1) ? bk : bv;
    float bias_c = bias[c];
#pragma unroll
    for (int i = 0; i < 8; ++i) acc[i] += bias_c;

    if (mat < 2) {
      bf16_t* dst = (mat == 0) ? qws : kws;
#pragma unroll
      for (int i = 0; i < 8; ++i) {
        int m = r * 16 + i + half * 8;
        dst[(size_t)(b * N_TOK + n0 + m) * C_DIM + c] = (bf16_t)acc[i];
      }
    } else {
      bf16x8 pk;
#pragma unroll
      for (int i = 0; i < 8; ++i) pk[i] = (bf16_t)acc[i];
      int mbase = r * 16 + half * 8;
      *(bf16x8*)(vtws + (size_t)(b * C_DIM + c) * N_TOK + n0 + mbase) = pk;
    }
  }
}

// ---------------------------------------------------------------- kernel 2
// flash attention, double-buffered async K/Vt staging
__global__ void __launch_bounds__(256)
attn_kernel(const float* __restrict__ x, const bf16_t* __restrict__ qws,
            const bf16_t* __restrict__ kws, const bf16_t* __restrict__ vtws,
            const float* __restrict__ gamma, float* __restrict__ out) {
  __shared__ bf16_t sQ[TILE * LDQ];
  __shared__ bf16_t sK[2][TILE * LDQ];
  __shared__ bf16_t sVt[2][C_DIM * LDVT];
  __shared__ float  sS[TILE * LD_S];
  __shared__ bf16_t sP[TILE * LDP];
  __shared__ float  sM[TILE], sL[TILE], sScale[TILE];
  __shared__ float  sRed[256];

  const int bid = blockIdx.x;
  const int b   = bid >> 6;
  const int n0  = (bid & 63) * TILE;
  const int tid  = threadIdx.x;
  const int lane = tid & 31;
  const int wave = tid >> 5;
  const int half = lane >> 4;
  const int ln   = lane & 15;

  const bf16_t* kbase  = kws  + (size_t)b * N_TOK * C_DIM;
  const bf16_t* vtbase = vtws + (size_t)b * C_DIM * N_TOK;

  // stage Q tile (stays resident)
  for (int idx = tid; idx < TILE * C_DIM; idx += 256) {
    int m = idx >> 8;
    int c = idx & 255;
    sQ[m * LDQ + c] = qws[(size_t)(b * N_TOK + n0 + m) * C_DIM + c];
  }
  if (tid < TILE) { sM[tid] = -3.0e38f; sL[tid] = 0.0f; }

  // prologue: issue async stage of tile 0 into buffer 0
  // (16 x b128 per thread: 8 for K rows, 8 for Vt rows)
#pragma unroll
  for (int it = 0; it < 8; ++it) {
    int idx = tid + it * 256;
    int m = idx >> 5, cc = (idx & 31) * 8;
    copy16_g2l(kbase + (size_t)m * C_DIM + cc, &sK[0][m * LDQ + cc]);
  }
#pragma unroll
  for (int it = 0; it < 8; ++it) {
    int idx = tid + it * 256;
    int c = idx >> 3, ii = (idx & 7) * 8;
    copy16_g2l(vtbase + (size_t)c * N_TOK + ii, &sVt[0][c * LDVT + ii]);
  }

  f32x8 acc[4][2];
#pragma unroll
  for (int rt = 0; rt < 4; ++rt) { acc[rt][0] = (f32x8){}; acc[rt][1] = (f32x8){}; }

  for (int kt = 0; kt < NT; ++kt) {
    const int buf = kt & 1;
    const bool more = (kt + 1) < NT;

    if (more) {   // issue next tile into the other buffer
      int k1 = (kt + 1) * TILE;
#pragma unroll
      for (int it = 0; it < 8; ++it) {
        int idx = tid + it * 256;
        int m = idx >> 5, cc = (idx & 31) * 8;
        copy16_g2l(kbase + (size_t)(k1 + m) * C_DIM + cc,
                   &sK[buf ^ 1][m * LDQ + cc]);
      }
#pragma unroll
      for (int it = 0; it < 8; ++it) {
        int idx = tid + it * 256;
        int c = idx >> 3, ii = (idx & 7) * 8;
        copy16_g2l(vtbase + (size_t)c * N_TOK + k1 + ii,
                   &sVt[buf ^ 1][c * LDVT + ii]);
      }
    }
    if (kt + 2 < NT) {  // L2 prefetch two tiles ahead (one 128B line/thread)
      int k2 = (kt + 2) * TILE;
      __builtin_prefetch(kbase + (size_t)k2 * C_DIM + (size_t)tid * 64, 0, 1);
      __builtin_prefetch(vtbase + (size_t)(tid) * N_TOK + k2, 0, 1);
    }
    wait_async_le16(more);   // current buffer's copies complete
    __syncthreads();

    // ---- S = Q K^T : 16 tiles of 16x16, two per wave
#pragma unroll
    for (int j = 0; j < 2; ++j) {
      int t  = wave * 2 + j;
      int rt = t >> 2, ct = t & 3;
      f32x8 s = {};
#pragma unroll
      for (int kk = 0; kk < 8; ++kk) {
        bf16x16 a  = load_fragA(sQ, rt * 16 + ln, LDQ, kk * 32, lane);
        bf16x16 bb = load_fragB(sK[buf], ct * 16 + ln, LDQ, kk * 32, lane);
        s = wmma_bf16(a, bb, s);
      }
#pragma unroll
      for (int i = 0; i < 8; ++i)
        sS[(rt * 16 + i + half * 8) * LD_S + ct * 16 + ln] = s[i];
    }
    __syncthreads();

    // ---- online softmax, 4 threads per row
    {
      int row = tid >> 2, q = tid & 3;
      const float* Sr = &sS[row * LD_S + q * 16];
      float mx = -3.0e38f;
#pragma unroll
      for (int jj = 0; jj < 16; ++jj) mx = fmaxf(mx, Sr[jj]);
      sRed[tid] = mx;
    }
    __syncthreads();
    if (tid < TILE) {
      float mx = fmaxf(fmaxf(sRed[tid * 4], sRed[tid * 4 + 1]),
                       fmaxf(sRed[tid * 4 + 2], sRed[tid * 4 + 3]));
      float mold = sM[tid];
      float mnew = fmaxf(mold, mx);
      sScale[tid] = __expf(mold - mnew);
      sM[tid] = mnew;
    }
    __syncthreads();
    {
      int row = tid >> 2, q = tid & 3;
      float mnew = sM[row];
      const float* Sr = &sS[row * LD_S + q * 16];
      bf16_t* Pr = &sP[row * LDP + q * 16];
      float sum = 0.0f;
#pragma unroll
      for (int jj = 0; jj < 16; ++jj) {
        float p = __expf(Sr[jj] - mnew);
        Pr[jj] = (bf16_t)p;
        sum += p;
      }
      sRed[tid] = sum;
    }
    __syncthreads();
    if (tid < TILE) {
      sL[tid] = sL[tid] * sScale[tid] +
                (sRed[tid * 4] + sRed[tid * 4 + 1] +
                 sRed[tid * 4 + 2] + sRed[tid * 4 + 3]);
    }

    // ---- rescale O, accumulate P @ V (wave owns channel tiles 2w, 2w+1)
#pragma unroll
    for (int rt = 0; rt < 4; ++rt) {
      float sc[8];
#pragma unroll
      for (int i = 0; i < 8; ++i) sc[i] = sScale[rt * 16 + i + half * 8];
#pragma unroll
      for (int j = 0; j < 2; ++j) {
#pragma unroll
        for (int i = 0; i < 8; ++i) acc[rt][j][i] *= sc[i];
        int c0 = (wave * 2 + j) * 16;
#pragma unroll
        for (int kk = 0; kk < 2; ++kk) {
          bf16x16 a  = load_fragA(sP, rt * 16 + ln, LDP, kk * 32, lane);
          bf16x16 bb = load_fragB(sVt[buf], c0 + ln, LDVT, kk * 32, lane);
          acc[rt][j] = wmma_bf16(a, bb, acc[rt][j]);
        }
      }
    }
    __syncthreads();
  }

  // ---- epilogue: out = gamma * (O / l) + x, in [B][C][H*W] layout
  float g = gamma[0];
#pragma unroll
  for (int rt = 0; rt < 4; ++rt) {
#pragma unroll
    for (int j = 0; j < 2; ++j) {
      int c = (wave * 2 + j) * 16 + ln;
#pragma unroll
      for (int i = 0; i < 8; ++i) {
        int m = rt * 16 + i + half * 8;
        float val = acc[rt][j][i] / sL[m];
        size_t addr = (size_t)(b * C_DIM + c) * N_TOK + n0 + m;
        out[addr] = g * val + x[addr];
      }
    }
  }
}

extern "C" void kernel_launch(void* const* d_in, const int* in_sizes, int n_in,
                              void* d_out, int out_size, void* d_ws, size_t ws_size,
                              hipStream_t stream) {
  const float* x     = (const float*)d_in[0];
  const float* Wq    = (const float*)d_in[1];
  const float* bq    = (const float*)d_in[2];
  const float* Wk    = (const float*)d_in[3];
  const float* bk    = (const float*)d_in[4];
  const float* Wv    = (const float*)d_in[5];
  const float* bv    = (const float*)d_in[6];
  const float* gamma = (const float*)d_in[7];
  float* out = (float*)d_out;

  bf16_t* wAll = (bf16_t*)d_ws;                          // 3*C*C bf16
  bf16_t* qws  = wAll + (size_t)3 * C_DIM * C_DIM;       // [B][N][C]
  bf16_t* kws  = qws + (size_t)B_SZ * N_TOK * C_DIM;     // [B][N][C]
  bf16_t* vtws = kws + (size_t)B_SZ * N_TOK * C_DIM;     // [B][C][N]

  prep_weights<<<(3 * C_DIM * C_DIM + 255) / 256, 256, 0, stream>>>(Wq, Wk, Wv, wAll);
  qkv_kernel<<<B_SZ * 64, 256, 0, stream>>>(x, wAll, bq, bk, bv, qws, kws, vtws);
  attn_kernel<<<B_SZ * 64, 256, 0, stream>>>(x, qws, kws, vtws, gamma, out);
}
